// Attention_1005022347695
// MI455X (gfx1250) — matrix operand order repaired and checked by reading, not compiled, not run on hardware
//
#include <hip/hip_runtime.h>
#include <cstdint>
#include <cstddef>

// ---------- types ----------
typedef __attribute__((ext_vector_type(16))) __bf16 v16bf;
typedef __attribute__((ext_vector_type(8)))  float  v8f;
typedef __attribute__((ext_vector_type(2)))  float  v2f;
typedef __attribute__((ext_vector_type(4)))  float  f32x4;
typedef __attribute__((ext_vector_type(4)))  __bf16 bf16x4;
typedef __attribute__((ext_vector_type(4)))  unsigned int u32x4;

#define B_  8
#define Q_  16
#define KV_ 2048
#define D_  512   // NQ = NK = NV = H = 512

union BFFrag { v16bf v; u32x4 q[2]; };

__device__ __forceinline__ float fast_tanh(float x) {
#if __has_builtin(__builtin_amdgcn_tanhf)
  return __builtin_amdgcn_tanhf(x);
#else
  return tanhf(x);
#endif
}
__device__ __forceinline__ float fast_exp2(float x) {
#if __has_builtin(__builtin_amdgcn_exp2f)
  return __builtin_amdgcn_exp2f(x);
#else
  return exp2f(x);
#endif
}

// ---------- 1) f32 -> bf16 conversion (vectorized 4-wide) ----------
__global__ void cvt_f32_to_bf16(const float* __restrict__ in,
                                __bf16* __restrict__ out, int n4) {
  int i = blockIdx.x * blockDim.x + threadIdx.x;
  if (i < n4) {
    f32x4 v = *(const f32x4*)(in + 4 * (size_t)i);
    bf16x4 o;
    o.x = (__bf16)v.x; o.y = (__bf16)v.y; o.z = (__bf16)v.z; o.w = (__bf16)v.w;
    *(bf16x4*)(out + 4 * (size_t)i) = o;
  }
}

// ---------- 2) split W (512 x 1024) into Wq / Wk bf16 (512 x 512 each) ----------
__global__ void split_w_bf16(const float* __restrict__ W,
                             __bf16* __restrict__ wq, __bf16* __restrict__ wk) {
  int h = blockIdx.x;
  for (int k = threadIdx.x; k < D_; k += blockDim.x) {
    wq[(size_t)h * D_ + k] = (__bf16)W[(size_t)h * 1024 + k];
    wk[(size_t)h * D_ + k] = (__bf16)W[(size_t)h * 1024 + D_ + k];
  }
}

// ---------- 3) bf16 WMMA GEMM: C[m][h] = sum_k A[m][k] * Wrow[h][k] ----------
// A: M x 512 bf16 (K-major rows), Bm: 512 x 512 bf16 (K-major rows), C: M x 512 f32
// Workgroup tile 128(M) x 256(N), 8 waves (2M x 4N), each wave 64x64:
// 16 WMMAs per k-step against 8 fragment loads -> ~28 flop/byte from L2.
__global__ __launch_bounds__(256) void gemm_bf16(const __bf16* __restrict__ A,
                                                 const __bf16* __restrict__ Bm,
                                                 float* __restrict__ C, int M) {
  const int lane = threadIdx.x & 31;
  const int wave = threadIdx.x >> 5;
  const int m0 = blockIdx.x * 128 + (wave & 1) * 64;
  const int n0 = blockIdx.y * 256 + (wave >> 1) * 64;
  const int lr   = lane & 15;
  const int koff = (lane >> 4) * 8;    // A-frag: two 16B chunks at koff, 16+koff
  const int klo  = (lane >> 4) * 8;    // B-frag: two 16B chunks at klo, 16+klo

  v8f acc[4][4];
#pragma unroll
  for (int mi = 0; mi < 4; ++mi)
#pragma unroll
    for (int ni = 0; ni < 4; ++ni) {
      v8f z = {0.f,0.f,0.f,0.f,0.f,0.f,0.f,0.f};
      acc[mi][ni] = z;
    }

  for (int kk = 0; kk < D_; kk += 32) {
    BFFrag a[4], b[4];
#pragma unroll
    for (int mi = 0; mi < 4; ++mi) {
      const __bf16* pa = A + (size_t)(m0 + 16 * mi + lr) * D_ + kk + koff;
      a[mi].q[0] = *(const u32x4*)pa;
      a[mi].q[1] = *(const u32x4*)(pa + 16);
    }
#pragma unroll
    for (int ni = 0; ni < 4; ++ni) {
      const __bf16* pb = Bm + (size_t)(n0 + 16 * ni + lr) * D_ + kk + klo;
      b[ni].q[0] = *(const u32x4*)pb;
      b[ni].q[1] = *(const u32x4*)(pb + 16);
    }
#pragma unroll
    for (int mi = 0; mi < 4; ++mi)
#pragma unroll
      for (int ni = 0; ni < 4; ++ni)
        acc[mi][ni] = __builtin_amdgcn_wmma_f32_16x16x32_bf16(
            false, a[mi].v, false, b[ni].v, (short)0, acc[mi][ni], false, false);
  }

  const int mhi = 8 * (lane >> 4);
#pragma unroll
  for (int mi = 0; mi < 4; ++mi)
#pragma unroll
    for (int ni = 0; ni < 4; ++ni)
#pragma unroll
      for (int i = 0; i < 8; ++i) {
        int m = m0 + 16 * mi + i + mhi;
        int n = n0 + 16 * ni + lr;
        C[(size_t)m * D_ + n] = acc[mi][ni][i];
      }
}

// ---------- 4) energy + v-dot: e[b][q][kv] = v . tanh(kp + qp + bias) + vb ----------
// One wave per kv row; 8 waves/block; q_part tile + bias + v in LDS.
__global__ __launch_bounds__(256) void energy_kernel(
    const float* __restrict__ kp,   // [B*KV][512]
    const float* __restrict__ qp,   // [B*Q][512]
    const float* __restrict__ Wb,   // [512]
    const float* __restrict__ vw,   // [512]
    const float* __restrict__ vb,   // [1]
    float* __restrict__ e) {        // [B][Q][KV]
  __shared__ float s_qp[Q_ * D_];
  __shared__ float s_bias[D_];
  __shared__ float s_v[D_];
  const int b   = blockIdx.y;
  const int tid = threadIdx.x;
  for (int i = tid; i < Q_ * D_; i += 256) s_qp[i] = qp[(size_t)b * Q_ * D_ + i];
  for (int i = tid; i < D_; i += 256) { s_bias[i] = Wb[i]; s_v[i] = vw[i]; }
  __syncthreads();

  const int lane = tid & 31;
  const int wave = tid >> 5;
  const int row  = blockIdx.x * 8 + wave;   // kv index
  const float* kprow = kp + ((size_t)b * KV_ + row) * D_;
  float kreg[16];
#pragma unroll
  for (int j = 0; j < 16; ++j) kreg[j] = kprow[j * 32 + lane];
  const float vbias = vb[0];

  for (int q = 0; q < Q_; ++q) {
    float s = 0.f;
#pragma unroll
    for (int j = 0; j < 16; ++j) {
      int h = j * 32 + lane;
      s += fast_tanh(kreg[j] + s_qp[q * D_ + h] + s_bias[h]) * s_v[h];
    }
#pragma unroll
    for (int off = 16; off > 0; off >>= 1) s += __shfl_down(s, off, 32);
    if (lane == 0) e[((size_t)b * Q_ + q) * KV_ + row] = s + vbias;
  }
}

// ---------- 5) softmax over kv (in place), one block per (b,q) row ----------
__global__ __launch_bounds__(256) void softmax_kernel(float* __restrict__ e) {
  __shared__ float sred[8];
  __shared__ float sbc;
  float* p = e + (size_t)blockIdx.x * KV_;
  const int tid = threadIdx.x, lane = tid & 31, wave = tid >> 5;

  float vals[8];
  float m = -3.0e38f;
#pragma unroll
  for (int i = 0; i < 8; ++i) { vals[i] = p[tid + i * 256]; m = fmaxf(m, vals[i]); }
#pragma unroll
  for (int off = 16; off > 0; off >>= 1) m = fmaxf(m, __shfl_down(m, off, 32));
  if (lane == 0) sred[wave] = m;
  __syncthreads();
  if (tid == 0) {
    float mm = sred[0];
    for (int w = 1; w < 8; ++w) mm = fmaxf(mm, sred[w]);
    sbc = mm;
  }
  __syncthreads();
  m = sbc;
  __syncthreads();

  float s = 0.f;
#pragma unroll
  for (int i = 0; i < 8; ++i) {
    vals[i] = fast_exp2((vals[i] - m) * 1.44269504f);
    s += vals[i];
  }
#pragma unroll
  for (int off = 16; off > 0; off >>= 1) s += __shfl_down(s, off, 32);
  if (lane == 0) sred[wave] = s;
  __syncthreads();
  if (tid == 0) {
    float ss = 0.f;
    for (int w = 0; w < 8; ++w) ss += sred[w];
    sbc = ss;
  }
  __syncthreads();
  const float inv = 1.f / sbc;
#pragma unroll
  for (int i = 0; i < 8; ++i) p[tid + i * 256] = vals[i] * inv;
}

// ---------- 6) output: out[b][q][n] = sum_k values[b][k][n] * scores[b][q][k] ----------
// Full f32 precision via v_wmma_f32_16x16x4_f32; one wave per 16x16 output tile.
__global__ __launch_bounds__(256) void out_gemm(const float* __restrict__ values,
                                                const float* __restrict__ scores,
                                                float* __restrict__ out) {
  const int b    = blockIdx.y;
  const int lane = threadIdx.x & 31;
  const int wave = threadIdx.x >> 5;
  const int n0   = (blockIdx.x * 8 + wave) * 16;
  const int lr   = lane & 15;
  const int k2   = (lane >> 4) * 2;

  v8f acc = {0.f,0.f,0.f,0.f,0.f,0.f,0.f,0.f};
  const float* srow = scores + ((size_t)b * Q_ + lr) * KV_;       // A: row q = lr
  const float* vcol = values + (size_t)b * KV_ * D_ + n0 + lr;    // B: column n

  for (int k = 0; k < KV_; k += 4) {
    v2f a, bb;
    const float2 av = *(const float2*)(srow + k + k2);
    a[0] = av.x; a[1] = av.y;
    bb[0] = vcol[(size_t)(k + k2) * D_];
    bb[1] = vcol[(size_t)(k + k2 + 1) * D_];
    acc = __builtin_amdgcn_wmma_f32_16x16x4_f32(false, a, false, bb,
                                                (short)0, acc, false, false);
  }
#pragma unroll
  for (int i = 0; i < 8; ++i) {
    int q = i + 8 * (lane >> 4);
    out[((size_t)b * Q_ + q) * D_ + n0 + lr] = acc[i];
  }
}

// ---------- host ----------
extern "C" void kernel_launch(void* const* d_in, const int* in_sizes, int n_in,
                              void* d_out, int out_size, void* d_ws, size_t ws_size,
                              hipStream_t stream) {
  const float* query  = (const float*)d_in[0];   // 8*16*512
  const float* keys   = (const float*)d_in[1];   // 8*2048*512
  const float* values = (const float*)d_in[2];   // 8*2048*512
  const float* Ww     = (const float*)d_in[3];   // 512*1024
  const float* Wb     = (const float*)d_in[4];   // 512
  const float* vw     = (const float*)d_in[5];   // 512
  const float* vb     = (const float*)d_in[6];   // 1
  float* out = (float*)d_out;

  char* ws = (char*)d_ws;
  size_t off = 0;
  auto carve = [&](size_t bytes) {
    void* p = ws + off;
    off += (bytes + 255) & ~(size_t)255;
    return p;
  };
  __bf16* keys_bf  = (__bf16*)carve((size_t)B_ * KV_ * D_ * 2);  // 16.8 MB
  __bf16* query_bf = (__bf16*)carve((size_t)B_ * Q_ * D_ * 2);   // 128 KB
  __bf16* wq_bf    = (__bf16*)carve((size_t)D_ * D_ * 2);        // 512 KB
  __bf16* wk_bf    = (__bf16*)carve((size_t)D_ * D_ * 2);        // 512 KB
  float*  kp       = (float*)carve((size_t)B_ * KV_ * D_ * 4);   // 33.6 MB
  float*  qp       = (float*)carve((size_t)B_ * Q_ * D_ * 4);    // 256 KB
  float*  e        = (float*)carve((size_t)B_ * Q_ * KV_ * 4);   // 1 MB

  // 1) conversions
  {
    int n4 = (B_ * KV_ * D_) / 4;
    cvt_f32_to_bf16<<<(n4 + 255) / 256, 256, 0, stream>>>(keys, keys_bf, n4);
  }
  {
    int n4 = (B_ * Q_ * D_) / 4;
    cvt_f32_to_bf16<<<(n4 + 255) / 256, 256, 0, stream>>>(query, query_bf, n4);
  }
  split_w_bf16<<<D_, 256, 0, stream>>>(Ww, wq_bf, wk_bf);

  // 2) projections via bf16 WMMA (wg tile 128x256, wave tile 64x64)
  gemm_bf16<<<dim3((B_ * KV_) / 128, D_ / 256), 256, 0, stream>>>(keys_bf, wk_bf, kp, B_ * KV_);
  gemm_bf16<<<dim3((B_ * Q_) / 128, D_ / 256), 256, 0, stream>>>(query_bf, wq_bf, qp, B_ * Q_);

  // 3) energy (hardware tanh) + v-dot
  energy_kernel<<<dim3(KV_ / 8, B_), 256, 0, stream>>>(kp, qp, Wb, vw, vb, e);

  // 4) softmax over kv
  softmax_kernel<<<B_ * Q_, 256, 0, stream>>>(e);

  // 5) output GEMM in full f32 via v_wmma_f32_16x16x4_f32
  out_gemm<<<dim3(D_ / (16 * 8), B_), 256, 0, stream>>>(values, e, out);
}
